// CombinedMLPMoEModel_38912403702040
// MI455X (gfx1250) — matrix operand order repaired and checked by reading, not compiled, not run on hardware
//
#include <hip/hip_runtime.h>
#include <hip/hip_bf16.h>

// ---------------------------------------------------------------------------
// Types
// ---------------------------------------------------------------------------
typedef __attribute__((ext_vector_type(16))) __bf16 v16bf;
typedef __attribute__((ext_vector_type(8)))  __bf16 v8bf;
typedef __attribute__((ext_vector_type(4)))  __bf16 v4bf;
typedef __attribute__((ext_vector_type(8)))  float  v8f;
typedef __attribute__((ext_vector_type(4)))  int    v4i;

#define BM 128
#define BN 128
#define BK 32
#define LDS_STRIDE 40   // (BK + 8) bf16 elems -> 80B row pitch, 16B aligned, conflict-reducing

// gfx1250 async global->LDS path (probe-verified on this toolchain)
#if defined(__has_builtin)
#if __has_builtin(__builtin_amdgcn_global_load_async_to_lds_b128) && \
    __has_builtin(__builtin_amdgcn_s_wait_asynccnt)
#define USE_ASYNC_LDS 1
#endif
#endif
#ifndef USE_ASYNC_LDS
#define USE_ASYNC_LDS 0
#endif

#define AS_GLOBAL __attribute__((address_space(1)))
#define AS_LDS    __attribute__((address_space(3)))

// Epilogue modes
//  0: out_bf16 = acc + bias
//  1: out_bf16 = relu(bn(acc + bias))
//  2: out_f32  = acc + bias
//  3: out_f32 += rowscale[row] * (acc + bias)     (MoE scaled accumulate)
//  4: out_f32  = bn(acc + bias)

template <int EPI>
__global__ __launch_bounds__(256)
void k_gemm_bf16(const __bf16* __restrict__ A, int lda,
                 const __bf16* __restrict__ B, int ldb,
                 float* __restrict__ outF, __bf16* __restrict__ outH, int ldc,
                 const float* __restrict__ bias,
                 const float* __restrict__ bng, const float* __restrict__ bnb,
                 const float* __restrict__ bnm, const float* __restrict__ bnv,
                 const float* __restrict__ rs, int rs_stride,
                 int M, int N, int K)
{
    __shared__ __align__(16) __bf16 As[2][BM * LDS_STRIDE];
    __shared__ __align__(16) __bf16 Bs[2][BN * LDS_STRIDE];
    __shared__ int s_any;

    const int tid  = threadIdx.x;
    const int lane = tid & 31;
    const int wid  = tid >> 5;          // 8 waves (wave32)
    const int wm   = wid >> 1;          // 0..3  -> 32-row slice
    const int wn   = wid & 1;           // 0..1  -> 64-col slice
    const int m0   = blockIdx.y * BM;
    const int n0   = blockIdx.x * BN;
    const int lh   = lane >> 4;         // K-half select per WMMA layout
    const int l16  = lane & 15;

    if (EPI == 3) {
        // Skip tiles where no row routed to this expert.
        if (tid == 0) s_any = 0;
        __syncthreads();
        if (tid < BM) {
            if (rs[(size_t)(m0 + tid) * rs_stride] != 0.0f) s_any = 1;
        }
        __syncthreads();
        if (!s_any) return;
    }

    v8f acc[2][4];
#pragma unroll
    for (int tm = 0; tm < 2; ++tm)
#pragma unroll
        for (int tn = 0; tn < 4; ++tn) acc[tm][tn] = (v8f){};

    // ---- staging helpers --------------------------------------------------
    auto issueA = [&](int k0, int b) {
#if USE_ASYNC_LDS
        // Async DMA global -> LDS (ASYNCcnt tracked, no VGPR round-trip).
#pragma unroll
        for (int it = 0; it < 2; ++it) {
            int t  = tid + it * 256;     // 0..511
            int r  = t >> 2;             // row 0..127
            int cc = t & 3;              // 8-elem chunk
            __builtin_amdgcn_global_load_async_to_lds_b128(
                (AS_GLOBAL v4i*)(A + (size_t)(m0 + r) * lda + k0 + cc * 8),
                (AS_LDS v4i*)&As[b][r * LDS_STRIDE + cc * 8],
                0, 0);
        }
#else
#pragma unroll
        for (int it = 0; it < 2; ++it) {
            int t  = tid + it * 256;
            int r  = t >> 2;
            int cc = t & 3;
            *(v8bf*)&As[b][r * LDS_STRIDE + cc * 8] =
                *(const v8bf*)(A + (size_t)(m0 + r) * lda + k0 + cc * 8);
        }
#endif
    };
    auto loadB = [&](int k0, v8bf* breg) {
#pragma unroll
        for (int it = 0; it < 2; ++it) {
            int t  = tid + it * 256;
            int kr = t >> 4;             // 0..31
            int cc = t & 15;             // 8-col chunk
            breg[it] = *(const v8bf*)(B + (size_t)(k0 + kr) * ldb + n0 + cc * 8);
        }
    };
    auto storeB = [&](const v8bf* breg, int b) {
#pragma unroll
        for (int it = 0; it < 2; ++it) {
            int t  = tid + it * 256;
            int kr = t >> 4;
            int cc = t & 15;
#pragma unroll
            for (int i = 0; i < 8; ++i)
                Bs[b][(cc * 8 + i) * LDS_STRIDE + kr] = breg[it][i];
        }
    };
    auto compute = [&](int b) {
        // ISA 16-bit A 16x32 / B 32x16 fragment layouts.
        v16bf afr[2], bfr[4];
#pragma unroll
        for (int tm = 0; tm < 2; ++tm) {
            const __bf16* ap = &As[b][(wm * 32 + tm * 16 + l16) * LDS_STRIDE + lh * 8];
            v8bf lo = *(const v8bf*)ap;
            v8bf hi = *(const v8bf*)(ap + 16);
            afr[tm] = __builtin_shufflevector(lo, hi, 0,1,2,3,4,5,6,7,8,9,10,11,12,13,14,15);
        }
#pragma unroll
        for (int tn = 0; tn < 4; ++tn) {
            const __bf16* bp = &Bs[b][(wn * 64 + tn * 16 + l16) * LDS_STRIDE + lh * 8];
            v8bf lo = *(const v8bf*)bp;
            v8bf hi = *(const v8bf*)(bp + 16);
            bfr[tn] = __builtin_shufflevector(lo, hi, 0,1,2,3,4,5,6,7,8,9,10,11,12,13,14,15);
        }
#pragma unroll
        for (int tm = 0; tm < 2; ++tm)
#pragma unroll
            for (int tn = 0; tn < 4; ++tn)
                acc[tm][tn] = __builtin_amdgcn_wmma_f32_16x16x32_bf16(
                    false, afr[tm], false, bfr[tn], (short)0, acc[tm][tn], false, false);
    };

    // ---- software-pipelined main loop (double-buffered LDS) ---------------
    v8bf breg[2];
    issueA(0, 0);
    loadB(0, breg);
    storeB(breg, 0);
#if USE_ASYNC_LDS
    __builtin_amdgcn_s_wait_asynccnt(0);
#endif
    __syncthreads();

    int buf = 0;
    for (int k0 = 0; k0 < K; k0 += BK) {
        const int  kn   = k0 + BK;
        const bool more = kn < K;
        if (more) {
            issueA(kn, buf ^ 1);        // async copy flies under the WMMAs
            loadB(kn, breg);            // global loads in flight during compute
        }
        compute(buf);
        if (more) {
            storeB(breg, buf ^ 1);
#if USE_ASYNC_LDS
            __builtin_amdgcn_s_wait_asynccnt(0);
#endif
        }
        __syncthreads();                // one barrier per k-step
        buf ^= 1;
    }

    // ---- epilogue ----
#pragma unroll
    for (int tm = 0; tm < 2; ++tm) {
        const int rbase = m0 + wm * 32 + tm * 16 + lh * 8;  // C layout: VGPR r -> M = r + 8*lh
        float rsv[8];
        if (EPI == 3) {
#pragma unroll
            for (int r = 0; r < 8; ++r)
                rsv[r] = rs[(size_t)(rbase + r) * rs_stride];
        }
#pragma unroll
        for (int tn = 0; tn < 4; ++tn) {
            const int col = n0 + wn * 64 + tn * 16 + l16;
            const float bc = bias[col];
            float scale = 0.0f, shift = 0.0f;
            if (EPI == 1 || EPI == 4) {
                const float s = bng[col] * rsqrtf(bnv[col] + 1e-5f);
                scale = s;
                shift = (bc - bnm[col]) * s + bnb[col];
            }
#pragma unroll
            for (int r = 0; r < 8; ++r) {
                const float v = acc[tm][tn][r];
                const size_t o = (size_t)(rbase + r) * ldc + col;
                if (EPI == 0) {
                    outH[o] = (__bf16)(v + bc);
                } else if (EPI == 1) {
                    float y = v * scale + shift;
                    outH[o] = (__bf16)(y > 0.0f ? y : 0.0f);
                } else if (EPI == 2) {
                    outF[o] = v + bc;
                } else if (EPI == 3) {
                    if (rsv[r] != 0.0f) outF[o] += rsv[r] * (v + bc);
                } else {
                    outF[o] = v * scale + shift;
                }
            }
        }
    }
}

// ---------------------------------------------------------------------------
// Helpers
// ---------------------------------------------------------------------------
__global__ void k_cvt_f32_bf16(const float* __restrict__ in, __bf16* __restrict__ out, long n)
{
    long i = ((long)blockIdx.x * 256 + threadIdx.x) * 4;
    if (i < n) {
        float4 v = *(const float4*)(in + i);
        v4bf o;
        o[0] = (__bf16)v.x; o[1] = (__bf16)v.y; o[2] = (__bf16)v.z; o[3] = (__bf16)v.w;
        *(v4bf*)(out + i) = o;
    }
}

__global__ void k_zero_f32(float* __restrict__ p, long n)
{
    long i = (long)blockIdx.x * 256 + threadIdx.x;
    if (i < n) p[i] = 0.0f;
}

__global__ __launch_bounds__(256)
void k_layernorm(const float* __restrict__ in, __bf16* __restrict__ out,
                 const float* __restrict__ g, const float* __restrict__ b, int cols)
{
    __shared__ float ss[256], sq[256];
    const int row = blockIdx.x;
    const float* x = in + (size_t)row * cols;
    float s = 0.0f, q = 0.0f;
    for (int j = threadIdx.x; j < cols; j += 256) {
        float v = x[j];
        s += v; q += v * v;
    }
    ss[threadIdx.x] = s; sq[threadIdx.x] = q;
    __syncthreads();
    for (int off = 128; off > 0; off >>= 1) {
        if (threadIdx.x < off) {
            ss[threadIdx.x] += ss[threadIdx.x + off];
            sq[threadIdx.x] += sq[threadIdx.x + off];
        }
        __syncthreads();
    }
    const float mean = ss[0] / cols;
    const float var  = sq[0] / cols - mean * mean;
    const float inv  = rsqrtf(var + 1e-5f);
    __bf16* o = out + (size_t)row * cols;
    for (int j = threadIdx.x; j < cols; j += 256)
        o[j] = (__bf16)((x[j] - mean) * inv * g[j] + b[j]);
}

// One wave32 per (row, chunk): 8 gate logits, top-2 softmax -> dense scale row.
__global__ __launch_bounds__(256)
void k_gate(const __bf16* __restrict__ m, int ldm,
            const float* __restrict__ gW, const float* __restrict__ gb,
            float* __restrict__ gsc)
{
    const int lane = threadIdx.x & 31;
    const int wid  = threadIdx.x >> 5;
    const int idx  = blockIdx.x * 8 + wid;       // 0 .. 4*8192-1
    const int c    = idx >> 13;                  // chunk
    const int row  = idx & 8191;
    const __bf16* x = m + (size_t)row * ldm + c * 1024;

    float acc[8] = {0,0,0,0,0,0,0,0};
    for (int j = lane; j < 1024; j += 32) {
        const float xv = (float)x[j];
        const float* wrow = gW + (size_t)j * 8;
#pragma unroll
        for (int e = 0; e < 8; ++e) acc[e] += xv * wrow[e];
    }
#pragma unroll
    for (int off = 16; off > 0; off >>= 1) {
#pragma unroll
        for (int e = 0; e < 8; ++e) acc[e] += __shfl_xor(acc[e], off, 32);
    }
    if (lane == 0) {
        float lg[8];
#pragma unroll
        for (int e = 0; e < 8; ++e) lg[e] = acc[e] + gb[e];
        int i0 = 0;
        for (int e = 1; e < 8; ++e) if (lg[e] > lg[i0]) i0 = e;
        int i1 = -1;
        for (int e = 0; e < 8; ++e) if (e != i0 && (i1 < 0 || lg[e] > lg[i1])) i1 = e;
        const float e1 = __expf(lg[i1] - lg[i0]);
        const float w1 = e1 / (1.0f + e1);
        const float w0 = 1.0f - w1;
        float* o = gsc + ((size_t)c * 8192 + row) * 8;
#pragma unroll
        for (int e = 0; e < 8; ++e) o[e] = 0.0f;
        o[i0] = w0;
        o[i1] = w1;
    }
}

// out[row] = dot(in[row,:], w) + rb  (one wave32 per row)
__global__ __launch_bounds__(256)
void k_rowdot(const float* __restrict__ in, const float* __restrict__ w,
              const float* __restrict__ rb, float* __restrict__ out, int cols)
{
    const int lane = threadIdx.x & 31;
    const int wid  = threadIdx.x >> 5;
    const int row  = blockIdx.x * 8 + wid;
    const float* x = in + (size_t)row * cols;
    float s = 0.0f;
    for (int j = lane; j < cols; j += 32) s += x[j] * w[j];
#pragma unroll
    for (int off = 16; off > 0; off >>= 1) s += __shfl_xor(s, off, 32);
    if (lane == 0) out[row] = s + rb[0];
}

// ---------------------------------------------------------------------------
// Host orchestration
// ---------------------------------------------------------------------------
extern "C" void kernel_launch(void* const* d_in, const int* in_sizes, int n_in,
                              void* d_out, int out_size, void* d_ws, size_t ws_size,
                              hipStream_t stream)
{
    (void)in_sizes; (void)n_in; (void)out_size; (void)ws_size;

    constexpr int B = 8192, D = 1024, H = 2048, IN = 2048, E = 8;
    constexpr int D4 = 4 * D;

    const float* x[4]  = {(const float*)d_in[0], (const float*)d_in[1],
                          (const float*)d_in[2], (const float*)d_in[3]};
    const float* pW[4] = {(const float*)d_in[4], (const float*)d_in[5],
                          (const float*)d_in[6], (const float*)d_in[7]};
    const float* pb[4] = {(const float*)d_in[8], (const float*)d_in[9],
                          (const float*)d_in[10], (const float*)d_in[11]};
    const float* mW1  = (const float*)d_in[12];
    const float* mb1  = (const float*)d_in[13];
    const float* bn1g = (const float*)d_in[14];
    const float* bn1b = (const float*)d_in[15];
    const float* bn1m = (const float*)d_in[16];
    const float* bn1v = (const float*)d_in[17];
    const float* mW2  = (const float*)d_in[18];
    const float* mb2  = (const float*)d_in[19];
    const float* bn2g = (const float*)d_in[20];
    const float* bn2b = (const float*)d_in[21];
    const float* bn2m = (const float*)d_in[22];
    const float* bn2v = (const float*)d_in[23];
    const float* mW3  = (const float*)d_in[24];
    const float* mb3  = (const float*)d_in[25];
    const float* lng  = (const float*)d_in[26];
    const float* lnb  = (const float*)d_in[27];
    const float* eW   = (const float*)d_in[28];
    const float* eb   = (const float*)d_in[29];
    const float* gW   = (const float*)d_in[30];
    const float* gb   = (const float*)d_in[31];
    const float* fW   = (const float*)d_in[32];
    const float* fb   = (const float*)d_in[33];
    const float* fbng = (const float*)d_in[34];
    const float* fbnb = (const float*)d_in[35];
    const float* fbnm = (const float*)d_in[36];
    const float* fbnv = (const float*)d_in[37];
    const float* rW   = (const float*)d_in[38];
    const float* rb   = (const float*)d_in[39];
    float* out = (float*)d_out;

    // ---- workspace layout (bytes) ----
    char* W = (char*)d_ws;
    const size_t MB = 1024ull * 1024ull;
    __bf16* pWb[4];
    for (int c = 0; c < 4; ++c) pWb[c] = (__bf16*)(W + c * 4 * MB);   // 4MB each
    __bf16* mW1b = (__bf16*)(W + 16 * MB);   // 16MB
    __bf16* mW2b = (__bf16*)(W + 32 * MB);   // 8MB
    __bf16* mW3b = (__bf16*)(W + 40 * MB);   // 16MB
    __bf16* eWb  = (__bf16*)(W + 56 * MB);   // 16MB
    __bf16* fWb  = (__bf16*)(W + 72 * MB);   // 8MB  -> weights end @80MB
    __bf16* xb[4];
    for (int c = 0; c < 4; ++c) xb[c] = (__bf16*)(W + 80 * MB + (size_t)c * 32 * MB); // 128MB
    float*  m_f32    = (float*)(W + 80 * MB);   // reuses xb region after projections
    float*  moe_f32  = m_f32;                   // reuses m_f32 after layernorm
    __bf16* combined = (__bf16*)(W + 208 * MB); // 64MB
    __bf16* fused_bf = combined;                // reuse after mW1 GEMM
    __bf16* h1       = (__bf16*)(W + 272 * MB); // 32MB
    float*  final_f  = (float*)(W + 272 * MB);  // reuse after mW2 GEMM
    __bf16* h2       = (__bf16*)(W + 304 * MB); // 32MB
    float*  gsc      = (float*)(W + 304 * MB);  // reuse after mW3 GEMM (1MB)
    __bf16* m_bf     = (__bf16*)(W + 336 * MB); // 64MB -> total 400MB

    auto cvt = [&](const float* src, __bf16* dst, size_t n) {
        dim3 g((unsigned)((n + 1023) / 1024));
        k_cvt_f32_bf16<<<g, 256, 0, stream>>>(src, dst, (long)n);
    };
    auto gemm = [&](int epi, const __bf16* A, int lda, const __bf16* Bm, int ldb,
                    float* oF, __bf16* oH, int ldc,
                    const float* bias, const float* g_, const float* b_,
                    const float* m_, const float* v_,
                    const float* rs, int rss, int M, int N, int K) {
        dim3 gr(N / BN, M / BM), bl(256);
        switch (epi) {
        case 0: k_gemm_bf16<0><<<gr, bl, 0, stream>>>(A, lda, Bm, ldb, oF, oH, ldc, bias, g_, b_, m_, v_, rs, rss, M, N, K); break;
        case 1: k_gemm_bf16<1><<<gr, bl, 0, stream>>>(A, lda, Bm, ldb, oF, oH, ldc, bias, g_, b_, m_, v_, rs, rss, M, N, K); break;
        case 2: k_gemm_bf16<2><<<gr, bl, 0, stream>>>(A, lda, Bm, ldb, oF, oH, ldc, bias, g_, b_, m_, v_, rs, rss, M, N, K); break;
        case 3: k_gemm_bf16<3><<<gr, bl, 0, stream>>>(A, lda, Bm, ldb, oF, oH, ldc, bias, g_, b_, m_, v_, rs, rss, M, N, K); break;
        default: k_gemm_bf16<4><<<gr, bl, 0, stream>>>(A, lda, Bm, ldb, oF, oH, ldc, bias, g_, b_, m_, v_, rs, rss, M, N, K); break;
        }
    };

    // 1) fp32 -> bf16 conversions (inputs + weights)
    for (int c = 0; c < 4; ++c) cvt(x[c], xb[c], (size_t)B * IN);
    for (int c = 0; c < 4; ++c) cvt(pW[c], pWb[c], (size_t)IN * D);
    cvt(mW1, mW1b, (size_t)D4 * H);
    cvt(mW2, mW2b, (size_t)H * H);
    cvt(mW3, mW3b, (size_t)H * D4);
    cvt(eW,  eWb,  (size_t)E * D * D);
    cvt(fW,  fWb,  (size_t)D4 * D);

    // 2) projections -> combined (bf16, columns c*D..)
    for (int c = 0; c < 4; ++c)
        gemm(0, xb[c], IN, pWb[c], D, nullptr, combined + c * D, D4,
             pb[c], nullptr, nullptr, nullptr, nullptr, nullptr, 0, B, D, IN);

    // 3) mW1 + BN1 + ReLU -> h1 (bf16)
    gemm(1, combined, D4, mW1b, H, nullptr, h1, H,
         mb1, bn1g, bn1b, bn1m, bn1v, nullptr, 0, B, H, D4);

    // 4) mW2 + BN2 + ReLU -> h2 (bf16)
    gemm(1, h1, H, mW2b, H, nullptr, h2, H,
         mb2, bn2g, bn2b, bn2m, bn2v, nullptr, 0, B, H, H);

    // 5) mW3 + bias -> m_f32
    gemm(2, h2, H, mW3b, D4, m_f32, nullptr, D4,
         mb3, nullptr, nullptr, nullptr, nullptr, nullptr, 0, B, D4, H);

    // 6) layernorm -> m_bf
    k_layernorm<<<dim3(B), 256, 0, stream>>>(m_f32, m_bf, lng, lnb, D4);

    // 7) gating: dense top-2 softmax scale matrix per chunk
    k_gate<<<dim3(4 * B / 8), 256, 0, stream>>>(m_bf, D4, gW, gb, gsc);

    // 8) MoE: zero accumulator, then scaled-accumulate GEMM per (chunk, expert)
    k_zero_f32<<<dim3((unsigned)(((size_t)B * D4 + 255) / 256)), 256, 0, stream>>>(
        moe_f32, (long)B * D4);
    for (int c = 0; c < 4; ++c)
        for (int e = 0; e < E; ++e)
            gemm(3, m_bf + c * D, D4, eWb + (size_t)e * D * D, D,
                 moe_f32 + c * D, nullptr, D4,
                 eb + (size_t)e * D, nullptr, nullptr, nullptr, nullptr,
                 gsc + (size_t)c * B * 8 + e, 8, B, D, D);

    // 9) fused (f32) -> bf16 for final GEMM
    cvt(moe_f32, fused_bf, (size_t)B * D4);

    // 10) fW + fb + final BN -> final_f (f32)
    gemm(4, fused_bf, D4, fWb, D, final_f, nullptr, D,
         fb, fbng, fbnb, fbnm, fbnv, nullptr, 0, B, D, D4);

    // 11) regression head: out = final_f @ rW + rb
    k_rowdot<<<dim3(B / 8), 256, 0, stream>>>(final_f, rW, rb, out, D);
}
